// MyAttention_5265629904983
// MI455X (gfx1250) — compile-verified, matrix operand order checked by reading, not run on hardware
//
#include <hip/hip_runtime.h>

typedef __attribute__((ext_vector_type(16))) _Float16 v16h;
typedef __attribute__((ext_vector_type(8)))  float    v8f;

#define N_   2
#define C_   128
#define H_   128
#define W_   128
#define HL   64
#define WL   64
#define HP   132          // H + 4   (halo 2, k/v rows)
#define WP   132          // W + 4   (halo 2, k cols)
#define WPV  144          // W + 4 + 12: padded K=32 agg loads stay in-bounds
#define HQ   130          // H + 2   (halo 1, q)
#define WQ   130
#define HW   (H_*W_)

// ---------------------------------------------------------------------------
// Workspace layout (bytes):
//   lr_up  f32 NCHW                      : [0,        16777216)
//   vpad   f16 [N][C][HP][WPV] (halo)    : [16777216, 26509312)
//   qpad   f16 [N][HQ][WQ][C]  (halo 1)  : [26509312, 35162112)
//   kpad   f16 [N][HP][WP][C]  (halo 2)  : [35162112, 44083200)
//   wa_wmma f16 [72][32][16] A-fragments : [44083200, 44156928)
// ---------------------------------------------------------------------------
#define OFF_VPAD  16777216u
#define OFF_QPAD  26509312u
#define OFF_KPAD  35162112u
#define OFF_WAW   44083200u
#define NZERO16   1706624

#define WMMA16(A, B, Cc) \
  __builtin_amdgcn_wmma_f32_16x16x32_f16(false, (A), false, (B), (short)0, (Cc), false, false)

__global__ void zero_fill(uint4* __restrict__ p, int n16) {
  int i = blockIdx.x * 256 + threadIdx.x;
  if (i < n16) p[i] = make_uint4(0u, 0u, 0u, 0u);
}

// thread = (n,c,y,x), x-minor. lr_up (bilinear align_corners) f32 NCHW and
// hr_value (dw3x3) as f16 into the haloed CHW value buffer.
__global__ void prep_nchw(const float* __restrict__ hr, const float* __restrict__ lr,
                          const float* __restrict__ wv, const float* __restrict__ bv,
                          float* __restrict__ lr_up, _Float16* __restrict__ vpad) {
  int idx = blockIdx.x * 256 + threadIdx.x;
  int x = idx & 127, y = (idx >> 7) & 127, c = (idx >> 14) & 127, n = idx >> 21;
  float fy = y * (63.0f / 127.0f);
  float fx = x * (63.0f / 127.0f);
  int y0 = (int)fy, x0 = (int)fx;
  int y1 = min(y0 + 1, HL - 1), x1 = min(x0 + 1, WL - 1);
  float wy = fy - (float)y0, wx = fx - (float)x0;
  const float* lrb = lr + (size_t)(n * C_ + c) * (HL * WL);
  float v00 = lrb[y0 * WL + x0], v01 = lrb[y0 * WL + x1];
  float v10 = lrb[y1 * WL + x0], v11 = lrb[y1 * WL + x1];
  lr_up[idx] = (v00 * (1.f - wx) + v01 * wx) * (1.f - wy)
             + (v10 * (1.f - wx) + v11 * wx) * wy;
  const float* hb = hr + (size_t)(n * C_ + c) * HW;
  float acc = bv[c];
#pragma unroll
  for (int ky = 0; ky < 3; ++ky) {
    int yy = y + ky - 1;
    if ((unsigned)yy < (unsigned)H_) {
#pragma unroll
      for (int kx = 0; kx < 3; ++kx) {
        int xx = x + kx - 1;
        if ((unsigned)xx < (unsigned)W_)
          acc += wv[c * 9 + ky * 3 + kx] * hb[yy * W_ + xx];
      }
    }
  }
  vpad[((size_t)(n * C_ + c) * HP + (y + 2)) * WPV + (x + 2)] = (_Float16)acc;
}

// q = dw3x3(lr_up), k = dw3x3(hr): x-minor coalesced f32 reads, LDS transpose,
// channel-minor f16 stores into haloed NHWC buffers.
__global__ void prep_nhwc(const float* __restrict__ hr, const float* __restrict__ lr_up,
                          const float* __restrict__ wq, const float* __restrict__ bq,
                          const float* __restrict__ wk, const float* __restrict__ bk,
                          _Float16* __restrict__ qpad, _Float16* __restrict__ kpad) {
  __shared__ _Float16 sq[C_ * 32];
  __shared__ _Float16 sk[C_ * 32];
  int b = blockIdx.x;
  int xt = b & 3, y = (b >> 2) & 127, n = b >> 9;
  int x0 = xt * 32;
  int t = threadIdx.x;
  int xl = t & 31, c0 = t >> 5;
  for (int i = 0; i < 16; ++i) {
    int c = c0 + 8 * i;
    int x = x0 + xl;
    const float* hb = hr + (size_t)(n * C_ + c) * HW;
    const float* ub = lr_up + (size_t)(n * C_ + c) * HW;
    float ak = bk[c], aq = bq[c];
#pragma unroll
    for (int ky = 0; ky < 3; ++ky) {
      int yy = y + ky - 1;
      if ((unsigned)yy < (unsigned)H_) {
#pragma unroll
        for (int kx = 0; kx < 3; ++kx) {
          int xx = x + kx - 1;
          if ((unsigned)xx < (unsigned)W_) {
            ak += wk[c * 9 + ky * 3 + kx] * hb[yy * W_ + xx];
            aq += wq[c * 9 + ky * 3 + kx] * ub[yy * W_ + xx];
          }
        }
      }
    }
    sk[c * 32 + xl] = (_Float16)ak;
    sq[c * 32 + xl] = (_Float16)aq;
  }
  __syncthreads();
  for (int i = 0; i < 16; ++i) {
    int j = t + 256 * i;
    int c = j & 127, x2 = j >> 7;
    qpad[((size_t)(n * HQ + (y + 1)) * WQ + (x0 + x2 + 1)) * C_ + c] = sq[c * 32 + x2];
    kpad[((size_t)(n * HP + (y + 2)) * WP + (x0 + x2 + 2)) * C_ + c] = sk[c * 32 + x2];
  }
}

// Pack wa (2,256,3,3) into WMMA A fragments, K ordered tap-major:
// k = tap*256 + cc. Element (m,k): lane = (k%32>=16 ? 16:0)+m, half = k%16.
__global__ void wa_pack(const float* __restrict__ wa, _Float16* __restrict__ waw) {
  int idx = blockIdx.x * 256 + threadIdx.x;   // 72*512 = 36864
  int h = idx & 15, l = (idx >> 4) & 31, kk = idx >> 9;
  int m = l & 15, hi = l >> 4;
  int k = kk * 32 + hi * 16 + h;
  int tap = k >> 8, cc = k & 255;
  float v = (m < 2) ? wa[(m * 256 + cc) * 9 + tap] : 0.0f;
  waw[idx] = (_Float16)v;
}

// One wave per 16-pixel row segment. 152 WMMAs, restructured for ILP:
// every group of WMMAs loads all operands up-front from one base pointer
// (immediate offsets -> load clauses) and accumulates in multiple
// independent chains that are summed in VALU.
__global__ void fused_attn(const _Float16* __restrict__ qpad,
                           const _Float16* __restrict__ kpad,
                           const _Float16* __restrict__ vpad,
                           const _Float16* __restrict__ waw,
                           const float* __restrict__ ba,
                           const float* __restrict__ lr_up,
                           float* __restrict__ out) {
  __shared__ __align__(32) float    sS[5 * 2 * 16 * 16];   // sim tiles (10 KB)
  __shared__ __align__(32) _Float16 sA[5 * 32 * 16];       // agg A fragments (5 KB)
  __shared__ float sdyn[2 * 16];

  int b = blockIdx.x;
  int xt = b & 7, y = (b >> 3) & 127, n = b >> 10;
  int x0 = xt * 16;
  int l = threadIdx.x;
  int lo = l & 15, hi = l >> 4;

  // ---------- phase 1: sim (40 WMMAs, 4 chains per dy) ----------
  v16h aq[4];
  const _Float16* qp =
      qpad + ((size_t)(n * HQ + (y + 1)) * WQ + (x0 + lo + 1)) * C_ + hi * 16;
#pragma unroll
  for (int kk = 0; kk < 4; ++kk) aq[kk] = *(const v16h*)(qp + kk * 32);

#pragma unroll
  for (int dy = 0; dy < 5; ++dy) {              // padded key row = y + dy
    // t=0 tile base; t=1 tile is +4 pixels = +4*C_ halves
    const _Float16* kp =
        kpad + ((size_t)(n * HP + (y + dy)) * WP + (x0 + lo)) * C_ + hi * 16;
    v16h b0[4], b1[4];
#pragma unroll
    for (int kk = 0; kk < 4; ++kk) {
      b0[kk] = *(const v16h*)(kp + kk * 32);
      b1[kk] = *(const v16h*)(kp + 4 * C_ + kk * 32);
    }
    v8f a0 = {}, a1 = {}, a2 = {}, a3 = {};
    a0 = WMMA16(aq[0], b0[0], a0);
    a1 = WMMA16(aq[1], b0[1], a1);
    a2 = WMMA16(aq[0], b1[0], a2);
    a3 = WMMA16(aq[1], b1[1], a3);
    a0 = WMMA16(aq[2], b0[2], a0);
    a1 = WMMA16(aq[3], b0[3], a1);
    a2 = WMMA16(aq[2], b1[2], a2);
    a3 = WMMA16(aq[3], b1[3], a3);
    v8f t0 = a0 + a1, t1 = a2 + a3;
#pragma unroll
    for (int v = 0; v < 8; ++v) {
      sS[((dy * 2 + 0) * 16 + v + 8 * hi) * 16 + lo] = t0[v];
      sS[((dy * 2 + 1) * 16 + v + 8 * hi) * 16 + lo] = t1[v];
    }
  }

  // ---------- phase 2: gate conv GEMM (72 WMMAs, 4 chains) ----------
  // Group q8 = 4 K-steps sharing one tap and one buffer (even: q, odd: k).
  v8f g0 = {}, g1 = {}, g2 = {}, g3 = {};
#pragma unroll 2
  for (int q8 = 0; q8 < 18; ++q8) {
    int tap = q8 >> 1;                          // wave-uniform 0..8
    int ky = tap / 3, kx = tap - ky * 3;
    const _Float16* base;
    if ((q8 & 1) == 0)
      base = qpad + ((size_t)(n * HQ + (y + ky)) * WQ + (x0 + lo + kx)) * C_ + hi * 16;
    else
      base = kpad + ((size_t)(n * HP + (y + ky + 1)) * WP + (x0 + lo + kx + 1)) * C_
             + hi * 16;
    v16h bm0 = *(const v16h*)(base);
    v16h bm1 = *(const v16h*)(base + 32);
    v16h bm2 = *(const v16h*)(base + 64);
    v16h bm3 = *(const v16h*)(base + 96);
    const _Float16* wb = waw + (size_t)q8 * 2048 + l * 16;
    v16h aw0 = *(const v16h*)(wb);
    v16h aw1 = *(const v16h*)(wb + 512);
    v16h aw2 = *(const v16h*)(wb + 1024);
    v16h aw3 = *(const v16h*)(wb + 1536);
    g0 = WMMA16(aw0, bm0, g0);
    g1 = WMMA16(aw1, bm1, g1);
    g2 = WMMA16(aw2, bm2, g2);
    g3 = WMMA16(aw3, bm3, g3);
  }
  v8f gacc = (g0 + g1) + (g2 + g3);
  if (hi == 0) {                                // D rows 0,1 = the two gates
    sdyn[lo]      = 1.0f / (1.0f + __expf(-(gacc[0] + ba[0])));
    sdyn[16 + lo] = 1.0f / (1.0f + __expf(-(gacc[1] + ba[1])));
  }
  {                                             // zero A staging
    v16h z = {};
#pragma unroll
    for (int dy = 0; dy < 5; ++dy) *(v16h*)&sA[(dy * 32 + l) * 16] = z;
  }
  __syncthreads();

  // ---------- phase 3: softmax + build scaled banded A ----------
  if (l < 16) {
    int p = lo;
    float s[25];
    float mx = -1e30f;
#pragma unroll
    for (int dy = 0; dy < 5; ++dy)
#pragma unroll
      for (int j = 0; j < 5; ++j) {
        int colrel = p + j - 2;                 // [-2, 17]
        int tt = (colrel <= 13) ? 0 : 1;
        int nc = colrel + 2 - 4 * tt;           // [0, 15]
        float v = sS[((dy * 2 + tt) * 16 + p) * 16 + nc];
        s[dy * 5 + j] = v;
        mx = fmaxf(mx, v);
      }
    float sum = 0.f;
#pragma unroll
    for (int e = 0; e < 25; ++e) { s[e] = __expf(s[e] - mx); sum += s[e]; }
    float d0 = sdyn[p], d1 = sdyn[16 + p];
    float sc = d0 / sum;
#pragma unroll
    for (int e = 0; e < 25; ++e) s[e] *= sc;
    s[12] += d1;                                // dy=2, dx=2: center tap
#pragma unroll
    for (int dy = 0; dy < 5; ++dy)
#pragma unroll
      for (int j = 0; j < 5; ++j) {
        int k = p + j;                          // 0..19
        sA[(dy * 32 + ((k >> 4) * 16 + p)) * 16 + (k & 15)] = (_Float16)s[dy * 5 + j];
      }
  }
  __syncthreads();

  // ---------- phase 4: aggregation + epilogue (40 WMMAs) ----------
  v16h awd[5];
#pragma unroll
  for (int dy = 0; dy < 5; ++dy) awd[dy] = *(const v16h*)&sA[(dy * 32 + l) * 16];

#pragma unroll 2
  for (int ct = 0; ct < 8; ++ct) {
    int c = ct * 16 + lo;
    // 5 B fragments from one base pointer, row offsets = WPV halves
    const _Float16* vb =
        vpad + (((size_t)(n * C_ + c) * HP + y) * WPV + x0 + hi * 16);
    v16h vb0 = *(const v16h*)(vb);
    v16h vb1 = *(const v16h*)(vb + WPV);
    v16h vb2 = *(const v16h*)(vb + 2 * WPV);
    v16h vb3 = *(const v16h*)(vb + 3 * WPV);
    v16h vb4 = *(const v16h*)(vb + 4 * WPV);
    const float* up = lr_up + ((size_t)(n * C_ + c) * H_ + y) * W_ + x0 + 8 * hi;
    float4 u0 = *(const float4*)(up);
    float4 u1 = *(const float4*)(up + 4);
    v8f acc;
    acc[0] = u0.x; acc[1] = u0.y; acc[2] = u0.z; acc[3] = u0.w;
    acc[4] = u1.x; acc[5] = u1.y; acc[6] = u1.z; acc[7] = u1.w;
    v8f acc2 = {};
    acc  = WMMA16(awd[0], vb0, acc);            // chain A (depth 3, C = lr_up)
    acc2 = WMMA16(awd[3], vb3, acc2);           // chain B (depth 2, C = 0)
    acc  = WMMA16(awd[1], vb1, acc);
    acc2 = WMMA16(awd[4], vb4, acc2);
    acc  = WMMA16(awd[2], vb2, acc);
    v8f res = acc + acc2;
    float* op = out + ((size_t)(n * C_ + c) * H_ + y) * W_ + x0 + 8 * hi;
    *(float4*)op       = make_float4(res[0], res[1], res[2], res[3]);
    *(float4*)(op + 4) = make_float4(res[4], res[5], res[6], res[7]);
  }
}

extern "C" void kernel_launch(void* const* d_in, const int* in_sizes, int n_in,
                              void* d_out, int out_size, void* d_ws, size_t ws_size,
                              hipStream_t stream) {
  (void)in_sizes; (void)n_in; (void)out_size; (void)ws_size;
  const float* hr = (const float*)d_in[0];
  const float* lr = (const float*)d_in[1];
  const float* wq = (const float*)d_in[2];
  const float* bq = (const float*)d_in[3];
  const float* wk = (const float*)d_in[4];
  const float* bk = (const float*)d_in[5];
  const float* wv = (const float*)d_in[6];
  const float* bv = (const float*)d_in[7];
  const float* wa = (const float*)d_in[8];
  const float* ba = (const float*)d_in[9];
  float* out = (float*)d_out;

  char* ws = (char*)d_ws;
  float*    lr_up = (float*)(ws);
  _Float16* vpad  = (_Float16*)(ws + OFF_VPAD);
  _Float16* qpad  = (_Float16*)(ws + OFF_QPAD);
  _Float16* kpad  = (_Float16*)(ws + OFF_KPAD);
  _Float16* waw   = (_Float16*)(ws + OFF_WAW);

  zero_fill<<<(NZERO16 + 255) / 256, 256, 0, stream>>>((uint4*)(ws + OFF_VPAD), NZERO16);
  prep_nchw<<<(N_ * C_ * H_ * W_) / 256, 256, 0, stream>>>(hr, lr, wv, bv, lr_up, vpad);
  prep_nhwc<<<N_ * H_ * (W_ / 32), 256, 0, stream>>>(hr, lr_up, wq, bq, wk, bk, qpad, kpad);
  wa_pack<<<144, 256, 0, stream>>>(wa, waw);
  fused_attn<<<N_ * H_ * (W_ / 16), 32, 0, stream>>>(qpad, kpad, vpad, waw, ba, lr_up, out);
}
